// ChannelBlock_11072425689588
// MI455X (gfx1250) — matrix operand order repaired and checked by reading, not compiled, not run on hardware
//
#include <hip/hip_runtime.h>
#include <hip/hip_bf16.h>
#include <math.h>

typedef __bf16 bf16_t;
typedef __attribute__((ext_vector_type(16))) __bf16 v16bf;
typedef __attribute__((ext_vector_type(8)))  __bf16 v8bf;
typedef __attribute__((ext_vector_type(8)))  float  v8f;
typedef __attribute__((ext_vector_type(4)))  int    v4i;

#define BDIM 32
#define HH   28
#define WW   28
#define CDIM 384
#define NTOK 784                 // 28*28
#define MROWS (BDIM * NTOK)     // 25088
#define HEADS 8
#define HD    48
#define MLPH  1536

// CDNA5 async global->LDS path (ASYNCcnt), with portable fallback.
#if defined(__has_builtin)
#if __has_builtin(__builtin_amdgcn_global_load_async_to_lds_b128) && \
    __has_builtin(__builtin_amdgcn_s_wait_asynccnt)
#define USE_ASYNC_LDS 1
#endif
#endif

// ---------------------------------------------------------------------------
// f32 -> bf16 cast kernel (weights)
// ---------------------------------------------------------------------------
__global__ void cb_f2bf(const float* __restrict__ in, bf16_t* __restrict__ out,
                        size_t n) {
    size_t i = (size_t)blockIdx.x * blockDim.x + threadIdx.x;
    if (i < n) out[i] = (bf16_t)in[i];
}

// ---------------------------------------------------------------------------
// Depthwise 3x3 conv positional encoding: out = x + dwconv(x) + bias
// ---------------------------------------------------------------------------
__global__ void cb_dwconv(const float* __restrict__ x,
                          const float* __restrict__ w,   // (C,1,3,3)
                          const float* __restrict__ bias,
                          float* __restrict__ out) {
    size_t i = (size_t)blockIdx.x * blockDim.x + threadIdx.x;
    size_t total = (size_t)BDIM * NTOK * CDIM;
    if (i >= total) return;
    int c = (int)(i % CDIM);
    size_t rem = i / CDIM;
    int px = (int)(rem % WW); rem /= WW;
    int py = (int)(rem % HH);
    int b  = (int)(rem / HH);
    float acc = bias[c];
    #pragma unroll
    for (int ky = 0; ky < 3; ++ky) {
        int yy = py + ky - 1;
        if (yy < 0 || yy >= HH) continue;
        #pragma unroll
        for (int kx = 0; kx < 3; ++kx) {
            int xx = px + kx - 1;
            if (xx < 0 || xx >= WW) continue;
            acc += x[(((size_t)b * NTOK + yy * WW + xx) * CDIM) + c] *
                   w[c * 9 + ky * 3 + kx];
        }
    }
    out[i] = x[i] + acc;
}

// ---------------------------------------------------------------------------
// LayerNorm over C=384, one wave (32 lanes) per row, output bf16
// ---------------------------------------------------------------------------
__global__ __launch_bounds__(256) void cb_layernorm_bf16(
    const float* __restrict__ x, const float* __restrict__ g,
    const float* __restrict__ beta, bf16_t* __restrict__ out) {
    int lane = threadIdx.x & 31;
    int wave = threadIdx.x >> 5;
    int row  = blockIdx.x * 8 + wave;            // MROWS % 8 == 0
    const float* xr = x + (size_t)row * CDIM;
    float v[12];
    float s = 0.f;
    #pragma unroll
    for (int j = 0; j < 12; ++j) { v[j] = xr[lane + j * 32]; s += v[j]; }
    #pragma unroll
    for (int o = 16; o > 0; o >>= 1) s += __shfl_xor(s, o, 32);
    float mean = s * (1.f / CDIM);
    float q = 0.f;
    #pragma unroll
    for (int j = 0; j < 12; ++j) { float d = v[j] - mean; q += d * d; }
    #pragma unroll
    for (int o = 16; o > 0; o >>= 1) q += __shfl_xor(q, o, 32);
    float inv = rsqrtf(q * (1.f / CDIM) + 1e-5f);
    bf16_t* orow = out + (size_t)row * CDIM;
    #pragma unroll
    for (int j = 0; j < 12; ++j) {
        int c = lane + j * 32;
        orow[c] = (bf16_t)((v[j] - mean) * inv * g[c] + beta[c]);
    }
}

// ---------------------------------------------------------------------------
// bf16 WMMA GEMM: out[m,n] = sum_k A[m,k] * W[n,k]  (+bias, +gelu, +res)
// A: (M,K) bf16 row-major.  W: (N,K) bf16 row-major.
// Block tile 128x128, BK=32, 256 threads = 8 waves.
// Wave tile 64x32 -> 4x2 WMMA tiles = 8 v_wmma per K-step.
// Double-buffered LDS; global->LDS via async-to-LDS when available.
// ---------------------------------------------------------------------------
#define GBM 128
#define GBN 128
#define GBK 32
#define LDT 40   // halves per LDS tile row (32 + 8 pad; 80B stride, 16B aligned)

union FragBF { v16bf v; v8bf h[2]; };

__global__ __launch_bounds__(256) void cb_gemm_bf16(
    const bf16_t* __restrict__ A, const bf16_t* __restrict__ W,
    int M, int N, int K,
    const float* __restrict__ bias,   // nullable
    const float* __restrict__ res,    // nullable, (M,N) f32
    float* __restrict__ outf,         // f32 output (nullable)
    bf16_t* __restrict__ outh,        // bf16 output (nullable)
    int gelu) {
    __shared__ bf16_t As[2][GBM * LDT];
    __shared__ bf16_t Bs[2][GBN * LDT];

    int t    = threadIdx.x;
    int lane = t & 31;
    int wave = t >> 5;
    int wN   = (wave & 3) * 32;   // wave N-offset
    int wM   = (wave >> 2) * 64;  // wave M-offset
    int bm   = blockIdx.x;
    int bn   = blockIdx.y;

    v8f acc[4][2] = {};

    // Staging: row r = t>>1 (0..127) for both A and B, 16 halves per thread
    int r  = t >> 1;
    int co = (t & 1) * 16;
    const bf16_t* Ag = A + (size_t)(bm * GBM + r) * K + co;
    const bf16_t* Wg = W + (size_t)(bn * GBN + r) * K + co;
    int stoff = r * LDT + co;

    int mrow = lane & 15;
    int kbA  = (lane >> 4) * 8;   // A frag: halves 0..7 -> K=kbA..; 8..15 -> 16+kbA..
    int kbB  = (lane >> 4) * 8;   // B frag: halves 0..7 -> K=kbB..; 8..15 -> 16+kbB..

#define STAGE(buf, k0)                                                        \
    do {                                                                      \
        bf16_t* asd = &As[buf][stoff];                                        \
        bf16_t* bsd = &Bs[buf][stoff];                                        \
        STAGE16(asd, Ag + (k0));                                              \
        STAGE16(bsd, Wg + (k0));                                              \
    } while (0)

#ifdef USE_ASYNC_LDS
typedef __attribute__((address_space(1))) v4i gv4i_t;
typedef __attribute__((address_space(3))) v4i lv4i_t;
#define STAGE16(dst, src)                                                     \
    do {                                                                      \
        __builtin_amdgcn_global_load_async_to_lds_b128(                       \
            (gv4i_t*)(src), (lv4i_t*)(dst), 0, 0);                            \
        __builtin_amdgcn_global_load_async_to_lds_b128(                       \
            (gv4i_t*)((src) + 8), (lv4i_t*)((dst) + 8), 0, 0);                \
    } while (0)
#define ASYNC_WAIT() __builtin_amdgcn_s_wait_asynccnt(0)
#else
#define STAGE16(dst, src)                                                     \
    do {                                                                      \
        *(v8bf*)(dst)       = *(const v8bf*)(src);                            \
        *(v8bf*)((dst) + 8) = *(const v8bf*)((src) + 8);                      \
    } while (0)
#define ASYNC_WAIT() ((void)0)
#endif

    STAGE(0, 0);
    ASYNC_WAIT();
    __syncthreads();

    int buf = 0;
    for (int k0 = 0; k0 < K; k0 += GBK, buf ^= 1) {
        if (k0 + GBK < K) STAGE(buf ^ 1, k0 + GBK);

        FragBF af[4], bf[2];
        #pragma unroll
        for (int tm = 0; tm < 4; ++tm) {
            const bf16_t* ap = &As[buf][(wM + tm * 16 + mrow) * LDT];
            af[tm].h[0] = *(const v8bf*)(ap + kbA);
            af[tm].h[1] = *(const v8bf*)(ap + 16 + kbA);
        }
        #pragma unroll
        for (int tn = 0; tn < 2; ++tn) {
            const bf16_t* bp = &Bs[buf][(wN + tn * 16 + mrow) * LDT] + kbB;
            bf[tn].h[0] = *(const v8bf*)(bp);
            bf[tn].h[1] = *(const v8bf*)(bp + 16);
        }
        #pragma unroll
        for (int tm = 0; tm < 4; ++tm)
            #pragma unroll
            for (int tn = 0; tn < 2; ++tn)
                acc[tm][tn] = __builtin_amdgcn_wmma_f32_16x16x32_bf16(
                    false, af[tm].v, false, bf[tn].v,
                    (short)0, acc[tm][tn], false, false);

        ASYNC_WAIT();
        __syncthreads();
    }

    // Epilogue. C/D layout: VGPR r -> M = r + 8*(lane>>4); N = lane&15
    int mhalf = (lane >> 4) * 8;
    int nloc  = lane & 15;
    #pragma unroll
    for (int tm = 0; tm < 4; ++tm) {
        #pragma unroll
        for (int tn = 0; tn < 2; ++tn) {
            int gnb = bn * GBN + wN + tn * 16 + nloc;
            #pragma unroll
            for (int rr = 0; rr < 8; ++rr) {
                int gm = bm * GBM + wM + tm * 16 + mhalf + rr;
                size_t idx = (size_t)gm * N + gnb;
                float v = acc[tm][tn][rr];
                if (bias) v += bias[gnb];
                if (gelu) v = 0.5f * v * (1.f + erff(v * 0.70710678118f));
                if (res)  v += res[idx];
                if (outf) outf[idx] = v;
                else      outh[idx] = (bf16_t)v;
            }
        }
    }
}

// ---------------------------------------------------------------------------
// Channel attention, stage 1: S = softmax_e( scale * k^T v ) per (b, head)
// ---------------------------------------------------------------------------
__global__ __launch_bounds__(256) void cb_attn_kv(
    const bf16_t* __restrict__ qkv, float* __restrict__ attn) {
    int bh = blockIdx.x;
    int b  = bh >> 3;
    int h  = bh & 7;
    __shared__ bf16_t kbuf[16 * HD];
    __shared__ bf16_t vbuf[16 * HD];
    __shared__ float  S[HD * (HD + 1)];
    int t = threadIdx.x;
    float acc[9] = {0.f, 0.f, 0.f, 0.f, 0.f, 0.f, 0.f, 0.f, 0.f};

    for (int n0 = 0; n0 < NTOK; n0 += 16) {
        __syncthreads();
        for (int j = t; j < 16 * HD; j += 256) {
            int nn = j / HD, e = j % HD;
            size_t base = (size_t)(b * NTOK + n0 + nn) * (3 * CDIM);
            kbuf[j] = qkv[base + CDIM     + h * HD + e];
            vbuf[j] = qkv[base + 2 * CDIM + h * HD + e];
        }
        __syncthreads();
        #pragma unroll
        for (int j = 0; j < 9; ++j) {
            int p = t * 9 + j;
            int d = p / HD, e = p % HD;
            float a = acc[j];
            #pragma unroll
            for (int nn = 0; nn < 16; ++nn)
                a += (float)kbuf[nn * HD + d] * (float)vbuf[nn * HD + e];
            acc[j] = a;
        }
    }
    __syncthreads();
    const float scale = rsqrtf((float)HD);
    #pragma unroll
    for (int j = 0; j < 9; ++j) {
        int p = t * 9 + j;
        S[(p / HD) * (HD + 1) + (p % HD)] = acc[j] * scale;
    }
    __syncthreads();
    if (t < HD) {
        float* row = &S[t * (HD + 1)];
        float mx = -1e30f;
        for (int e = 0; e < HD; ++e) mx = fmaxf(mx, row[e]);
        float sum = 0.f;
        for (int e = 0; e < HD; ++e) { float ex = expf(row[e] - mx); row[e] = ex; sum += ex; }
        float inv = 1.f / sum;
        float* orow = attn + ((size_t)bh * HD + t) * HD;
        for (int e = 0; e < HD; ++e) orow[e] = row[e] * inv;
    }
}

// ---------------------------------------------------------------------------
// Channel attention, stage 2: out[m, h*48+d] = sum_e attn[bh,d,e] * q[m, h*48+e]
// ---------------------------------------------------------------------------
__global__ __launch_bounds__(128) void cb_attn_out(
    const bf16_t* __restrict__ qkv, const float* __restrict__ attn,
    bf16_t* __restrict__ out) {
    int m = blockIdx.x;
    int b = m / NTOK;
    __shared__ float qrow[CDIM];
    int t = threadIdx.x;
    for (int j = t; j < CDIM; j += 128)
        qrow[j] = (float)qkv[(size_t)m * (3 * CDIM) + j];
    __syncthreads();
    for (int j = t; j < CDIM; j += 128) {
        int h = j / HD, d = j % HD;
        const float* ar = attn + (((size_t)(b * HEADS + h) * HD) + d) * HD;
        const float* qr = qrow + h * HD;
        float s = 0.f;
        #pragma unroll
        for (int e = 0; e < HD; ++e) s += ar[e] * qr[e];
        out[(size_t)m * CDIM + j] = (bf16_t)s;
    }
}

// ---------------------------------------------------------------------------
extern "C" void kernel_launch(void* const* d_in, const int* in_sizes, int n_in,
                              void* d_out, int out_size, void* d_ws, size_t ws_size,
                              hipStream_t stream) {
    const float* x      = (const float*)d_in[0];
    const float* cpe0_w = (const float*)d_in[3];
    const float* cpe0_b = (const float*)d_in[4];
    const float* cpe1_w = (const float*)d_in[5];
    const float* cpe1_b = (const float*)d_in[6];
    const float* n1_g   = (const float*)d_in[7];
    const float* n1_b   = (const float*)d_in[8];
    const float* qkv_w  = (const float*)d_in[9];
    const float* proj_w = (const float*)d_in[10];
    const float* proj_b = (const float*)d_in[11];
    const float* n2_g   = (const float*)d_in[12];
    const float* n2_b   = (const float*)d_in[13];
    const float* fc1_w  = (const float*)d_in[14];
    const float* fc1_b  = (const float*)d_in[15];
    const float* fc2_w  = (const float*)d_in[16];
    const float* fc2_b  = (const float*)d_in[17];
    float* out = (float*)d_out;

    char* ws = (char*)d_ws;
    size_t off = 0;
    auto alloc = [&](size_t bytes) -> void* {
        void* p = ws + off;
        off = (off + bytes + 255) & ~(size_t)255;
        return p;
    };
    const size_t M = MROWS;
    float*  x1    = (float*) alloc(M * CDIM * 4);          // cpe0 out / x2
    bf16_t* qkvb  = (bf16_t*)alloc(M * 3 * CDIM * 2);      // qkv bf16
    bf16_t* actb  = (bf16_t*)alloc(M * CDIM * 2);          // ln1 / attnout / ln2
    float*  attnb = (float*) alloc((size_t)BDIM * HEADS * HD * HD * 4);
    bf16_t* hbuf  = (bf16_t*)alloc(M * MLPH * 2);          // mlp hidden bf16
    bf16_t* wq    = (bf16_t*)alloc((size_t)3 * CDIM * CDIM * 2);
    bf16_t* wp    = (bf16_t*)alloc((size_t)CDIM * CDIM * 2);
    bf16_t* w1    = (bf16_t*)alloc((size_t)MLPH * CDIM * 2);
    bf16_t* w2    = (bf16_t*)alloc((size_t)CDIM * MLPH * 2);
    (void)ws_size; (void)in_sizes; (void)n_in; (void)out_size;

    // Weight casts to bf16
    {
        size_t n;
        n = (size_t)3 * CDIM * CDIM;
        cb_f2bf<<<(unsigned)((n + 255) / 256), 256, 0, stream>>>(qkv_w, wq, n);
        n = (size_t)CDIM * CDIM;
        cb_f2bf<<<(unsigned)((n + 255) / 256), 256, 0, stream>>>(proj_w, wp, n);
        n = (size_t)MLPH * CDIM;
        cb_f2bf<<<(unsigned)((n + 255) / 256), 256, 0, stream>>>(fc1_w, w1, n);
        n = (size_t)CDIM * MLPH;
        cb_f2bf<<<(unsigned)((n + 255) / 256), 256, 0, stream>>>(fc2_w, w2, n);
    }

    size_t totalE = M * CDIM;
    unsigned convGrid = (unsigned)((totalE + 255) / 256);

    // x1 = conv_pos_enc(x)
    cb_dwconv<<<convGrid, 256, 0, stream>>>(x, cpe0_w, cpe0_b, x1);
    // actb = LN1(x1) as bf16
    cb_layernorm_bf16<<<(unsigned)(M / 8), 256, 0, stream>>>(x1, n1_g, n1_b, actb);
    // qkvb = actb @ qkv_w.T   (M x 1152)
    cb_gemm_bf16<<<dim3(M / GBM, (3 * CDIM) / GBN), 256, 0, stream>>>(
        actb, wq, (int)M, 3 * CDIM, CDIM, nullptr, nullptr, nullptr, qkvb, 0);
    // attention matrices + softmax
    cb_attn_kv<<<BDIM * HEADS, 256, 0, stream>>>(qkvb, attnb);
    // actb = attention output (bf16)
    cb_attn_out<<<(unsigned)M, 128, 0, stream>>>(qkvb, attnb, actb);
    // x1 = x1 + actb @ proj_w.T + proj_b   (in-place residual)
    cb_gemm_bf16<<<dim3(M / GBM, CDIM / GBN), 256, 0, stream>>>(
        actb, wp, (int)M, CDIM, CDIM, proj_b, x1, x1, nullptr, 0);
    // out = conv_pos_enc(x1)  (x3 lives in d_out)
    cb_dwconv<<<convGrid, 256, 0, stream>>>(x1, cpe1_w, cpe1_b, out);
    // actb = LN2(out) as bf16
    cb_layernorm_bf16<<<(unsigned)(M / 8), 256, 0, stream>>>(out, n2_g, n2_b, actb);
    // hbuf = gelu(actb @ fc1_w.T + fc1_b) as bf16
    cb_gemm_bf16<<<dim3(M / GBM, MLPH / GBN), 256, 0, stream>>>(
        actb, w1, (int)M, MLPH, CDIM, fc1_b, nullptr, nullptr, hbuf, 1);
    // out = out + hbuf @ fc2_w.T + fc2_b
    cb_gemm_bf16<<<dim3(M / GBM, CDIM / GBN), 256, 0, stream>>>(
        hbuf, w2, (int)M, CDIM, MLPH, fc2_b, out, out, nullptr, 0);
}
